// SingleRenderer_12335146074566
// MI455X (gfx1250) — matrix operand order repaired and checked by reading, not compiled, not run on hardware
//
#include <hip/hip_runtime.h>

typedef __attribute__((ext_vector_type(16))) _Float16 v16h;
typedef __attribute__((ext_vector_type(8)))  float    v8f;

union ABu { unsigned u[8]; v16h h; };

__device__ __forceinline__ float sigmoidf_(float x) { return 1.0f / (1.0f + __expf(-x)); }

// ---------------------------------------------------------------------------
// One-off: transpose + convert a 128x128 f32 weight matrix to f16 [n][k]
// ---------------------------------------------------------------------------
__global__ void k_prepw(const float* __restrict__ W, _Float16* __restrict__ WT)
{
    int idx = blockIdx.x * blockDim.x + threadIdx.x;
    if (idx >= 128 * 128) return;
    int n = idx >> 7, k = idx & 127;
    WT[n * 128 + k] = (_Float16)W[k * 128 + n];
}

// ---------------------------------------------------------------------------
// Fused 3-layer MLP: OUT = act3( relu( relu(X@W1+B1) @ W2 + B2 ) @ W3 + B3 )
// Block = 64 threads (2 waves); each wave owns a 32-point tile (two WMMA
// M-tiles), so every B fragment read from LDS feeds two WMMAs (halves LDS
// traffic per WMMA). A fragments for both tiles are register-resident.
// Layer2 (128x128) runs on v_wmma_f32_16x16x32_f16, f32 accumulate.
// W2T is pre-transposed f16 in global -> coalesced uint4 copy, conflict-free
// ds_store_b128 staging.
// ---------------------------------------------------------------------------
template<int IN_DIM, int OUT_DIM, bool SIG>
__global__ __launch_bounds__(64) void mlp_kernel(
    const float* __restrict__ X, float* __restrict__ OUT,
    const float* __restrict__ W1, const float* __restrict__ B1,
    const _Float16* __restrict__ W2T, const float* __restrict__ B2,
    const float* __restrict__ W3, const float* __restrict__ B3,
    int P)
{
    __shared__ __align__(16) _Float16 lw2t[128 * 128];   // W2^T [n][k] f16 (32 KB)
    __shared__ __align__(16) _Float16 lh[2][32 * 128];   // per-wave hidden (16 KB)
    __shared__ float lw1[IN_DIM * 128];
    __shared__ float lb1[128];
    __shared__ float lb2[128];
    __shared__ float lw3[128 * OUT_DIM];
    __shared__ float lb3[OUT_DIM];
    __shared__ float lx[2][32 * IN_DIM];

    const int tid  = threadIdx.x;
    const int wv   = tid >> 5;
    const int lane = tid & 31;
    const int M    = lane & 15;   // A row / B col / D col within a 16-tile
    const int hi   = lane >> 4;   // lane-half selector

    // Stage pre-transposed f16 W2 with vector copies (coalesced + conflict-free)
    {
        const uint4* src = (const uint4*)W2T;
        uint4*       dst = (uint4*)lw2t;
        for (int i = tid; i < (128 * 128) / 8; i += 64) dst[i] = src[i];
    }
    for (int i = tid; i < IN_DIM * 128; i += 64) lw1[i] = W1[i];
    for (int i = tid; i < 128; i += 64) { lb1[i] = B1[i]; lb2[i] = B2[i]; }
    for (int i = tid; i < 128 * OUT_DIM; i += 64) lw3[i] = W3[i];
    if (tid < OUT_DIM) lb3[tid] = B3[tid];

    const int p0 = blockIdx.x * 64 + wv * 32;   // 32 points per wave
    for (int i = lane; i < 32 * IN_DIM; i += 32) {
        int p = i / IN_DIM, k = i % IN_DIM;
        int gp = p0 + p;
        lx[wv][i] = (gp < P) ? X[gp * IN_DIM + k] : 0.0f;
    }
    __syncthreads();

    // ---- Layer 1 (skinny K): VALU, write h1 f16 to LDS --------------------
    for (int i = lane; i < 32 * 128; i += 32) {
        int p = i >> 7, j = i & 127;
        float acc = lb1[j];
        #pragma unroll
        for (int k = 0; k < IN_DIM; ++k)
            acc = fmaf(lx[wv][p * IN_DIM + k], lw1[k * 128 + j], acc);
        lh[wv][p * 128 + j] = (_Float16)fmaxf(acc, 0.0f);
    }
    __syncthreads();

    // ---- Layer 2: 32x128 @ 128x128 via WMMA -------------------------------
    // Cache A for both M-tiles, all 4 K-chunks, in registers (64 VGPRs).
    ABu a[2][4];
    #pragma unroll
    for (int mt = 0; mt < 2; ++mt)
        #pragma unroll
        for (int kc = 0; kc < 4; ++kc)
            #pragma unroll
            for (int d = 0; d < 8; ++d) {
                int ka = ((d >= 4) ? 16 : 0) + hi * 8 + 2 * (d & 3);
                a[mt][kc].u[d] =
                    *(const unsigned*)&lh[wv][(mt * 16 + M) * 128 + kc * 32 + ka];
            }

    // h1 now fully register-resident -> safe to overwrite lh with h2 per N-tile.
    #pragma unroll
    for (int nt = 0; nt < 8; ++nt) {
        v8f c0 = {}, c1 = {};
        #pragma unroll
        for (int kc = 0; kc < 4; ++kc) {
            ABu b;
            #pragma unroll
            for (int d = 0; d < 8; ++d) {
                int kb = ((d >= 4) ? 16 : 0) + hi * 8 + 2 * (d & 3);
                b.u[d] = *(const unsigned*)&lw2t[(nt * 16 + M) * 128 + kc * 32 + kb];
            }
            c0 = __builtin_amdgcn_wmma_f32_16x16x32_f16(
                     false, a[0][kc].h, false, b.h, (short)0, c0, false, false);
            c1 = __builtin_amdgcn_wmma_f32_16x16x32_f16(
                     false, a[1][kc].h, false, b.h, (short)0, c1, false, false);
        }
        #pragma unroll
        for (int r = 0; r < 8; ++r) {
            int m = r + hi * 8;
            int n = nt * 16 + M;
            float bias = lb2[n];
            lh[wv][(m)      * 128 + n] = (_Float16)fmaxf(c0[r] + bias, 0.0f);
            lh[wv][(m + 16) * 128 + n] = (_Float16)fmaxf(c1[r] + bias, 0.0f);
        }
    }
    __syncthreads();

    // ---- Layer 3 (skinny N): each lane owns one of the wave's 32 points ---
    float acc[OUT_DIM];
    #pragma unroll
    for (int o = 0; o < OUT_DIM; ++o) acc[o] = 0.0f;
    for (int k = 0; k < 128; ++k) {
        float h = (float)lh[wv][lane * 128 + k];
        #pragma unroll
        for (int o = 0; o < OUT_DIM; ++o)
            acc[o] = fmaf(h, lw3[k * OUT_DIM + o], acc[o]);
    }
    int gp = p0 + lane;
    if (gp < P) {
        #pragma unroll
        for (int o = 0; o < OUT_DIM; ++o) {
            float v = acc[o] + lb3[o];
            if (SIG) v = sigmoidf_(v);
            OUT[gp * OUT_DIM + o] = v;
        }
    }
}

// ---------------------------------------------------------------------------
// Per-ray helper kernels
// ---------------------------------------------------------------------------
__global__ void k_dirs(const float* __restrict__ rd, float* __restrict__ dirs, int R)
{
    int r = blockIdx.x * blockDim.x + threadIdx.x;
    if (r >= R) return;
    float x = rd[r * 3], y = rd[r * 3 + 1], z = rd[r * 3 + 2];
    float inv = rsqrtf(x * x + y * y + z * z);
    dirs[r * 3] = x * inv; dirs[r * 3 + 1] = y * inv; dirs[r * 3 + 2] = z * inv;
}

__global__ void k_coarse(const float* __restrict__ ro, const float* __restrict__ dirs,
                         float* __restrict__ dA, float* __restrict__ pts, int R)
{
    int idx = blockIdx.x * blockDim.x + threadIdx.x;
    if (idx >= R * 64) return;
    int r = idx >> 6, s = idx & 63;
    float t = (float)s / 63.0f;
    float d = 0.1f * (1.0f - t) + 2.0f * t;
    dA[r * 128 + s] = d;
    pts[idx * 3 + 0] = ro[r * 3 + 0] + dirs[r * 3 + 0] * d;
    pts[idx * 3 + 1] = ro[r * 3 + 1] + dirs[r * 3 + 1] * d;
    pts[idx * 3 + 2] = ro[r * 3 + 2] + dirs[r * 3 + 2] * d;
}

__global__ void k_copy_coarse(const float* __restrict__ sTmp, float* __restrict__ sA, int R)
{
    int idx = blockIdx.x * blockDim.x + threadIdx.x;
    if (idx >= R * 64) return;
    sA[(idx >> 6) * 128 + (idx & 63)] = sTmp[idx];
}

__global__ void k_importance(const float* __restrict__ dcur, const float* __restrict__ scur,
                             const float* __restrict__ ro, const float* __restrict__ dirs,
                             float* __restrict__ dF, float* __restrict__ ptsF,
                             int n, int iter, int R)
{
    int r = blockIdx.x * blockDim.x + threadIdx.x;
    if (r >= R) return;
    const float* dv = dcur + r * 128;
    const float* sv = scur + r * 128;
    float s_i = 64.0f * (float)(1 << iter);

    float cdf[129];
    cdf[0] = 0.0f;
    float T = 1.0f, prev_dot = 0.0f, wsum = 0.0f;
    for (int j = 0; j < n - 1; ++j) {
        float ps = sv[j], ns = sv[j + 1];
        float pz = dv[j], nz = dv[j + 1];
        float mid = 0.5f * (ps + ns);
        float dot = (ns - ps) / (nz - pz + 1e-5f);
        float dc = fminf(prev_dot, dot);
        prev_dot = dot;
        dc = fminf(fmaxf(dc, -10.0f), 0.0f);
        float dist = nz - pz;
        float pc = sigmoidf_((mid - dc * dist * 0.5f) * s_i);
        float nc = sigmoidf_((mid + dc * dist * 0.5f) * s_i);
        float alpha = (pc - nc + 1e-5f) / (pc + 1e-5f);
        float w = alpha * T;                  // alpha_to_w
        T *= (1.0f - alpha + 1e-10f);
        w += 1e-5f;
        wsum += w;
        cdf[j + 1] = wsum;
    }
    float inv = 1.0f / wsum;
    for (int j = 1; j < n; ++j) cdf[j] *= inv;

    const float* o = ro + r * 3;
    const float* v = dirs + r * 3;
    for (int q = 0; q < 16; ++q) {
        float u = (float)q / 15.0f;
        int lo = 0, hiI = n;                  // searchsorted right on cdf[0..n-1]
        while (lo < hiI) {
            int m = (lo + hiI) >> 1;
            if (cdf[m] <= u) lo = m + 1; else hiI = m;
        }
        int below = lo - 1; if (below < 0) below = 0; if (below > n - 1) below = n - 1;
        int above = lo;     if (above > n - 1) above = n - 1;
        float cb = cdf[below], ca = cdf[above];
        float denom = ca - cb;
        if (denom < 1e-5f) denom = 1.0f;
        float t = (u - cb) / denom;
        float dq = dv[below] + t * (dv[above] - dv[below]);
        dF[r * 16 + q] = dq;
        ptsF[(r * 16 + q) * 3 + 0] = o[0] + v[0] * dq;
        ptsF[(r * 16 + q) * 3 + 1] = o[1] + v[1] * dq;
        ptsF[(r * 16 + q) * 3 + 2] = o[2] + v[2] * dq;
    }
}

// Merge two sorted (d,sdf) lists (argsort on concat == stable merge; fine ones sorted)
__global__ void k_merge(const float* __restrict__ dA, const float* __restrict__ sA,
                        const float* __restrict__ dF, const float* __restrict__ sF,
                        float* __restrict__ dB, float* __restrict__ sB, int n, int R)
{
    int r = blockIdx.x * blockDim.x + threadIdx.x;
    if (r >= R) return;
    const float* da = dA + r * 128; const float* sa = sA + r * 128;
    const float* df = dF + r * 16;  const float* sf = sF + r * 16;
    float* db = dB + r * 128;       float* sb = sB + r * 128;
    int i = 0, j = 0;
    for (int k = 0; k < n + 16; ++k) {
        bool takeA = (j >= 16) || (i < n && da[i] <= df[j]);
        if (takeA) { db[k] = da[i]; sb[k] = sa[i]; ++i; }
        else       { db[k] = df[j]; sb[k] = sf[j]; ++j; }
    }
}

__global__ void k_prepfinal(const float* __restrict__ dcur,
                            const float* __restrict__ ro, const float* __restrict__ dirs,
                            float* __restrict__ pts6, int R)
{
    int idx = blockIdx.x * blockDim.x + threadIdx.x;
    if (idx >= R * 127) return;
    int r = idx / 127, i = idx % 127;
    float dm = 0.5f * (dcur[r * 128 + i] + dcur[r * 128 + i + 1]);
    float* p = pts6 + idx * 6;
    p[0] = ro[r * 3 + 0] + dirs[r * 3 + 0] * dm;
    p[1] = ro[r * 3 + 1] + dirs[r * 3 + 1] * dm;
    p[2] = ro[r * 3 + 2] + dirs[r * 3 + 2] * dm;
    p[3] = dirs[r * 3 + 0]; p[4] = dirs[r * 3 + 1]; p[5] = dirs[r * 3 + 2];
}

__global__ void k_composite(const float* __restrict__ dcur, const float* __restrict__ scur,
                            const float* __restrict__ rad, const float* __restrict__ s_param,
                            float* __restrict__ out, int R)
{
    int r = blockIdx.x * blockDim.x + threadIdx.x;
    if (r >= R) return;
    float s = __expf(10.0f * s_param[0]);
    const float* dv = dcur + r * 128;
    const float* sv = scur + r * 128;
    float T = 1.0f, r0 = 0, r1 = 0, r2 = 0, dep = 0, acc = 0;
    float c_prev = sigmoidf_(sv[0] * s);
    for (int i = 0; i < 127; ++i) {
        float c_next = sigmoidf_(sv[i + 1] * s);
        float alpha = fmaxf((c_prev - c_next) / (c_prev + 1e-10f), 0.0f);
        float w = alpha * T;
        T *= (1.0f - alpha + 1e-10f);
        float dm = 0.5f * (dv[i] + dv[i + 1]);
        const float* rr = rad + (r * 127 + i) * 3;
        r0 += w * rr[0]; r1 += w * rr[1]; r2 += w * rr[2];
        dep += w * dm; acc += w;
        c_prev = c_next;
    }
    out[r * 3 + 0] = r0; out[r * 3 + 1] = r1; out[r * 3 + 2] = r2;
    out[R * 3 + r] = dep / (acc + 1e-10f);
    out[R * 4 + r] = acc;
}

// ---------------------------------------------------------------------------
extern "C" void kernel_launch(void* const* d_in, const int* in_sizes, int n_in,
                              void* d_out, int out_size, void* d_ws, size_t ws_size,
                              hipStream_t stream)
{
    (void)in_sizes; (void)n_in; (void)out_size; (void)ws_size;
    const float* rays_o = (const float*)d_in[0];
    const float* rays_d = (const float*)d_in[1];
    const float* w1  = (const float*)d_in[2];
    const float* b1  = (const float*)d_in[3];
    const float* w2  = (const float*)d_in[4];
    const float* b2  = (const float*)d_in[5];
    const float* w3  = (const float*)d_in[6];
    const float* b3  = (const float*)d_in[7];
    const float* r1  = (const float*)d_in[8];
    const float* rb1 = (const float*)d_in[9];
    const float* r2  = (const float*)d_in[10];
    const float* rb2 = (const float*)d_in[11];
    const float* r3  = (const float*)d_in[12];
    const float* rb3 = (const float*)d_in[13];
    const float* s_param = (const float*)d_in[14];

    const int R = 4096;
    float* p    = (float*)d_ws;
    float* dirs = p; p += R * 3;
    float* dA   = p; p += R * 128;
    float* dB   = p; p += R * 128;
    float* sA   = p; p += R * 128;
    float* sB   = p; p += R * 128;
    float* sTmp = p; p += R * 64;
    float* dF   = p; p += R * 16;
    float* sF   = p; p += R * 16;
    float* pts  = p; p += R * 127 * 6;   // reused: coarse 3ch, fine 3ch, mid 6ch
    float* rad  = p; p += R * 127 * 3;
    _Float16* w2t = (_Float16*)p; p += (128 * 128) / 2;  // f16 transposed W2
    _Float16* r2t = (_Float16*)p; p += (128 * 128) / 2;  // f16 transposed r2

    // One-off weight transpose/convert for the WMMA B operand
    k_prepw<<<(128 * 128) / 256, 256, 0, stream>>>(w2, w2t);
    k_prepw<<<(128 * 128) / 256, 256, 0, stream>>>(r2, r2t);

    k_dirs<<<R / 256, 256, 0, stream>>>(rays_d, dirs, R);
    k_coarse<<<(R * 64) / 256, 256, 0, stream>>>(rays_o, dirs, dA, pts, R);
    mlp_kernel<3, 1, false><<<(R * 64) / 64, 64, 0, stream>>>(
        pts, sTmp, w1, b1, w2t, b2, w3, b3, R * 64);
    k_copy_coarse<<<(R * 64) / 256, 256, 0, stream>>>(sTmp, sA, R);

    float* dcur = dA; float* scur = sA; float* dnx = dB; float* snx = sB;
    int n = 64;
    for (int it = 0; it < 4; ++it) {
        k_importance<<<R / 256, 256, 0, stream>>>(dcur, scur, rays_o, dirs, dF, pts, n, it, R);
        mlp_kernel<3, 1, false><<<(R * 16) / 64, 64, 0, stream>>>(
            pts, sF, w1, b1, w2t, b2, w3, b3, R * 16);
        k_merge<<<R / 256, 256, 0, stream>>>(dcur, scur, dF, sF, dnx, snx, n, R);
        float* td = dcur; dcur = dnx; dnx = td;
        float* ts = scur; scur = snx; snx = ts;
        n += 16;
    }

    k_prepfinal<<<(R * 127 + 255) / 256, 256, 0, stream>>>(dcur, rays_o, dirs, pts, R);
    mlp_kernel<6, 3, true><<<(R * 127) / 64, 64, 0, stream>>>(
        pts, rad, r1, rb1, r2t, rb2, r3, rb3, R * 127);
    k_composite<<<R / 256, 256, 0, stream>>>(dcur, scur, rad, s_param, (float*)d_out, R);
}
